// ProblematicGeneAttentionLayer_21345987461805
// MI455X (gfx1250) — matrix operand order repaired and checked by reading, not compiled, not run on hardware
//
#include <hip/hip_runtime.h>

typedef __bf16 bf16_t;
typedef __attribute__((ext_vector_type(16))) __bf16 v16bf;
typedef __attribute__((ext_vector_type(8)))  __bf16 v8bf;
typedef __attribute__((ext_vector_type(8)))  float  v8f;

#define WMMA_BF16(A,B,C) \
  __builtin_amdgcn_wmma_f32_16x16x32_bf16(false,(A),false,(B),(short)0,(C),false,false)

#define BB 4
#define SS 1024
#define DD 512
#define HH 8
#define HD 64

// One async 16-byte copy global -> LDS (CDNA5 ASYNCcnt path).
// VDST = 32-bit LDS byte address, VADDR = 64-bit global address.
static __device__ __forceinline__ void async_copy16(unsigned lds_addr,
                                                    const bf16_t* gsrc) {
  asm volatile("global_load_async_to_lds_b128 %0, %1, off"
               :: "v"(lds_addr), "v"(gsrc) : "memory");
}
#define S_WAIT_ASYNCCNT(n) asm volatile("s_wait_asynccnt " #n ::: "memory")

// Assemble a 16-bit A-matrix fragment from two 8-element groups
// (contiguous k order: halves 0-7 = K colBase..+7, halves 8-15 = K colBase+8..+15)
static __device__ __forceinline__ v16bf ld2x8(const bf16_t* p0, const bf16_t* p1) {
  v8bf a = *(const v8bf*)p0;
  v8bf b = *(const v8bf*)p1;
  v16bf r;
#pragma unroll
  for (int i = 0; i < 8; ++i) { r[i] = a[i]; r[i + 8] = b[i]; }
  return r;
}

// ---------------------------------------------------------------------------
// Kernel 0: V [b,s,h,hd] f32 -> Vt [b,h,hd,s] bf16 (key-dim contiguous so PV
// B-tiles and the async LDS staging are contiguous 16B segments)
// ---------------------------------------------------------------------------
__global__ void k_vt(const float* __restrict__ V, bf16_t* __restrict__ Vt) {
  int i = blockIdx.x * 256 + threadIdx.x;     // dest index over B*H*HD*S = 2M
  int s  = i & (SS - 1);
  int t  = i >> 10;
  int hd = t & (HD - 1); t >>= 6;
  int h  = t & (HH - 1);
  int b  = t >> 3;
  Vt[i] = (bf16_t)V[(((size_t)b * SS + s) * HH + h) * HD + hd];
}

// ---------------------------------------------------------------------------
// Kernel 1: QK projection. y[4096,1024] = x[4096,512] @ Wqk^T + b, bf16-WMMA,
// f32 accumulate. Q pre-scaled by 1/sqrt(64) (bias folded), stored [b,h,s,64];
// K stored [b,h,s,64] (key-major = WMMA B layout for the scores matmul).
// ---------------------------------------------------------------------------
__global__ __launch_bounds__(32)
void k_qkproj(const float* __restrict__ X, const float* __restrict__ W,
              const float* __restrict__ bias,
              bf16_t* __restrict__ Qb, bf16_t* __restrict__ Kb) {
  const int lane = threadIdx.x & 31;
  const int g = lane >> 4, c = lane & 15;
  const int mt = blockIdx.y, nt = blockIdx.x;
  const float* xrow = X + (size_t)(mt * 16 + c) * DD;
  const float* wrow = W + (size_t)(nt * 16 + c) * DD;
  v8f acc = {0.f, 0.f, 0.f, 0.f, 0.f, 0.f, 0.f, 0.f};
#pragma unroll 4
  for (int kb = 0; kb < DD; kb += 32) {
    v16bf a;
    const float* pa0 = xrow + kb + g * 16;
    const float* pa1 = pa0 + 8;
#pragma unroll
    for (int i = 0; i < 8; ++i) { a[i] = (bf16_t)pa0[i]; a[i + 8] = (bf16_t)pa1[i]; }
    v16bf bm;
    const float* pb = wrow + kb + g * 16;
#pragma unroll
    for (int i = 0; i < 16; ++i) bm[i] = (bf16_t)pb[i];
    acc = WMMA_BF16(a, bm, acc);
  }
  const int ncol = nt * 16 + c;
  const float scale = (ncol < DD) ? 0.125f : 1.0f;
  const float bv = bias[ncol];
  bf16_t* dst;
  int n2;
  if (ncol < DD) { dst = Qb; n2 = ncol; } else { dst = Kb; n2 = ncol - DD; }
  const int h = n2 >> 6, hd = n2 & 63;
#pragma unroll
  for (int r = 0; r < 8; ++r) {
    int mrow = mt * 16 + r + g * 8;
    int b_ = mrow >> 10, s_ = mrow & (SS - 1);
    dst[(((size_t)b_ * HH + h) * SS + s_) * HD + hd] = (bf16_t)((acc[r] + bv) * scale);
  }
}

// ---------------------------------------------------------------------------
// Kernel 2: fused masked attention. 4 waves / block, 64 queries per (b,h).
// K/V chunks (4KB each) double-buffered in LDS via global_load_async_to_lds,
// overlapped with WMMA on the previous chunk (ASYNCcnt + barrier protocol).
// Online masked softmax: A_bar = e*M / (sum(e*M) + EPS*sum(e)); running max
// reduced cross-lane each chunk, sums kept as per-lane partials and reduced
// once at the end. P repacked C->A layout through wave-private LDS.
// ---------------------------------------------------------------------------
__global__ __launch_bounds__(128)
void k_attn(const bf16_t* __restrict__ Q, const bf16_t* __restrict__ K,
            const bf16_t* __restrict__ Vt, const float* __restrict__ Mm,
            bf16_t* __restrict__ AO) {
  __shared__ __align__(32) bf16_t Ks[2][32 * 64];   // [key][hd]   4KB x2
  __shared__ __align__(32) bf16_t Vs[2][64 * 32];   // [hd][key]   4KB x2
  __shared__ __align__(32) bf16_t Ps[4][16 * 32];   // per-wave P  1KB x4

  const int tid = threadIdx.x;
  const int w = tid >> 5, lane = tid & 31;
  const int g = lane >> 4, c = lane & 15;
  const int b = blockIdx.z, h = blockIdx.y;
  const int qb = blockIdx.x * 64 + w * 16;
  const size_t bh = (size_t)(b * HH + h);

  const unsigned ksA[2] = {(unsigned)(uintptr_t)&Ks[0][0], (unsigned)(uintptr_t)&Ks[1][0]};
  const unsigned vsA[2] = {(unsigned)(uintptr_t)&Vs[0][0], (unsigned)(uintptr_t)&Vs[1][0]};

  // cooperative async stage of one 32-key chunk into buffer `buf`
  auto issue = [&](int kc, int buf) {
    const bf16_t* gK = K + (bh * SS + kc) * HD;     // 4KB fully contiguous
    unsigned kdst = ksA[buf] + (unsigned)tid * 16;
    async_copy16(kdst,        gK + tid * 8);
    async_copy16(kdst + 2048, gK + tid * 8 + 1024);
    // V chunk: 64 rows x 64B, each row = 4 segments of 16B
    int r0 = tid >> 2, c0 = tid & 3;                // segments 0..127
    async_copy16(vsA[buf] + (unsigned)(r0 * 32 + c0 * 8) * 2,
                 Vt + (bh * HD + r0) * SS + kc + c0 * 8);
    int r1 = (tid + 128) >> 2, c1 = tid & 3;        // segments 128..255
    async_copy16(vsA[buf] + (unsigned)(r1 * 32 + c1 * 8) * 2,
                 Vt + (bh * HD + r1) * SS + kc + c1 * 8);
  };

  // per-wave Q A-tiles (hd 0..31, 32..63)
  const bf16_t* qrow = Q + (bh * SS + qb + c) * HD;
  const v16bf qa0 = ld2x8(qrow + g * 16,      qrow + g * 16 + 8);
  const v16bf qa1 = ld2x8(qrow + 32 + g * 16, qrow + 32 + g * 16 + 8);

  const float* Mp = Mm + ((size_t)b * SS + qb) * SS;
  bf16_t* myP = &Ps[w][0];

  float mrun[8], zf[8], zm[8];
  v8f o[4];
#pragma unroll
  for (int r = 0; r < 8; ++r) { mrun[r] = -1e30f; zf[r] = 0.f; zm[r] = 0.f; }
#pragma unroll
  for (int j = 0; j < 4; ++j)
#pragma unroll
    for (int r = 0; r < 8; ++r) o[j][r] = 0.f;

  issue(0, 0);
  int p = 0;
  for (int kc = 0; kc < SS; kc += 32) {
    if (kc + 32 < SS) {                 // prefetch next chunk, wait for current
      issue(kc + 32, p ^ 1);
      S_WAIT_ASYNCCNT(4);
    } else {
      S_WAIT_ASYNCCNT(0);
    }
    __syncthreads();

    const bf16_t* kS = &Ks[p][0];
    const bf16_t* vS = &Vs[p][0];

    // ---- scores: two 16x16 tiles, 2 WMMAs each over hd=64 (B from LDS) ----
    v16bf kb00 = *(const v16bf*)(kS + (c)      * 64 + g * 16);
    v16bf kb01 = *(const v16bf*)(kS + (c)      * 64 + 32 + g * 16);
    v16bf kb10 = *(const v16bf*)(kS + (16 + c) * 64 + g * 16);
    v16bf kb11 = *(const v16bf*)(kS + (16 + c) * 64 + 32 + g * 16);
    v8f s0 = {0.f, 0.f, 0.f, 0.f, 0.f, 0.f, 0.f, 0.f};
    v8f s1 = {0.f, 0.f, 0.f, 0.f, 0.f, 0.f, 0.f, 0.f};
    s0 = WMMA_BF16(qa0, kb00, s0);
    s0 = WMMA_BF16(qa1, kb01, s0);
    s1 = WMMA_BF16(qa0, kb10, s1);
    s1 = WMMA_BF16(qa1, kb11, s1);

    // ---- biological mask (C-layout addressing) ----
    float mk0[8], mk1[8];
#pragma unroll
    for (int r = 0; r < 8; ++r) {
      const float* mr = Mp + (size_t)(r + g * 8) * SS + kc + c;
      mk0[r] = mr[0];
      mk1[r] = mr[16];
    }
    if (kc + 32 < SS)
      __builtin_prefetch(Mp + (size_t)(g * 8) * SS + kc + 32 + c, 0, 1);

    // ---- online masked softmax: only the MAX needs a cross-lane reduce ----
#pragma unroll
    for (int r = 0; r < 8; ++r) {
      float sr0 = s0[r], sr1 = s1[r];
      float rmax = fmaxf(sr0, sr1);
#pragma unroll
      for (int d = 1; d < 16; d <<= 1) rmax = fmaxf(rmax, __shfl_xor(rmax, d, 16));
      float mnew = fmaxf(mrun[r], rmax);
      float corr = __expf(mrun[r] - mnew);
      mrun[r] = mnew;
      zf[r] *= corr; zm[r] *= corr;
      o[0][r] *= corr; o[1][r] *= corr; o[2][r] *= corr; o[3][r] *= corr;
      float p0 = __expf(sr0 - mnew), p1 = __expf(sr1 - mnew);
      float pm0 = p0 * mk0[r],       pm1 = p1 * mk1[r];
      zf[r] += p0 + p1;          // per-lane partial sums (reduced at the end)
      zm[r] += pm0 + pm1;
      const int row = r + g * 8;
      myP[row * 32 + c]      = (bf16_t)pm0;
      myP[row * 32 + 16 + c] = (bf16_t)pm1;
    }

    // ---- P(16x32) x V(32x64): P in A-layout from LDS, 4 PV WMMAs ----
    v16bf pa = ld2x8(myP + c * 32 + g * 16, myP + c * 32 + g * 16 + 8);
#pragma unroll
    for (int j = 0; j < 4; ++j) {
      v16bf vb = *(const v16bf*)(vS + (j * 16 + c) * 32 + g * 16);
      o[j] = WMMA_BF16(pa, vb, o[j]);
    }
    __syncthreads();            // protect K/V buffer before next async write
    p ^= 1;
  }

  // ---- final cross-lane sum reductions + normalize; store [b,q,h,hd] ----
#pragma unroll
  for (int r = 0; r < 8; ++r) {
    float zfr = zf[r], zmr = zm[r];
#pragma unroll
    for (int d = 1; d < 16; d <<= 1) {
      zfr += __shfl_xor(zfr, d, 16);
      zmr += __shfl_xor(zmr, d, 16);
    }
    float inv = 1.0f / (zmr + 1e-8f * zfr);
    int row = qb + r + g * 8;
#pragma unroll
    for (int j = 0; j < 4; ++j)
      AO[(((size_t)b * SS + row) * HH + h) * HD + j * 16 + c] = (bf16_t)(o[j][r] * inv);
  }
}

// ---------------------------------------------------------------------------
// Kernel 3: output projection. out[4096,512] = AO(bf16) @ out_w^T + out_b
// ---------------------------------------------------------------------------
__global__ __launch_bounds__(32)
void k_oproj(const bf16_t* __restrict__ AO, const float* __restrict__ W,
             const float* __restrict__ bias, float* __restrict__ out) {
  const int lane = threadIdx.x & 31;
  const int g = lane >> 4, c = lane & 15;
  const int mt = blockIdx.y, nt = blockIdx.x;
  const bf16_t* arow = AO + (size_t)(mt * 16 + c) * DD;
  const float*  wrow = W  + (size_t)(nt * 16 + c) * DD;
  v8f acc = {0.f, 0.f, 0.f, 0.f, 0.f, 0.f, 0.f, 0.f};
#pragma unroll 4
  for (int kb = 0; kb < DD; kb += 32) {
    v16bf a = ld2x8(arow + kb + g * 16, arow + kb + g * 16 + 8);
    v16bf bm;
    const float* pb = wrow + kb + g * 16;
#pragma unroll
    for (int i = 0; i < 16; ++i) bm[i] = (bf16_t)pb[i];
    acc = WMMA_BF16(a, bm, acc);
  }
  const float bv = bias[nt * 16 + c];
#pragma unroll
  for (int r = 0; r < 8; ++r)
    out[(size_t)(mt * 16 + r + g * 8) * DD + nt * 16 + c] = acc[r] + bv;
}

// ---------------------------------------------------------------------------
extern "C" void kernel_launch(void* const* d_in, const int* in_sizes, int n_in,
                              void* d_out, int out_size, void* d_ws, size_t ws_size,
                              hipStream_t stream) {
  const float* x   = (const float*)d_in[0];   // [4,1024,512]
  const float* V   = (const float*)d_in[1];   // [4,1024,8,64]
  const float* M   = (const float*)d_in[2];   // [4,1024,1024]
  const float* ipw = (const float*)d_in[3];   // [1536,512]
  const float* ipb = (const float*)d_in[4];   // [1536]
  const float* ow  = (const float*)d_in[5];   // [512,512]
  const float* ob  = (const float*)d_in[6];   // [512]
  float* out = (float*)d_out;                 // [4,1024,512]

  char* ws = (char*)d_ws;
  const size_t SEG = (size_t)4 << 20;
  bf16_t* Qb = (bf16_t*)(ws + 0 * SEG);       // [b,h,s,64]
  bf16_t* Kb = (bf16_t*)(ws + 1 * SEG);       // [b,h,s,64]
  bf16_t* Vt = (bf16_t*)(ws + 2 * SEG);       // [b,h,hd,s]
  bf16_t* AO = (bf16_t*)(ws + 3 * SEG);       // [b,q,h,hd] == [4096,512]

  k_vt    <<<dim3(8192),       256, 0, stream>>>(V, Vt);
  k_qkproj<<<dim3(64, 256),     32, 0, stream>>>(x, ipw, ipb, Qb, Kb);
  k_attn  <<<dim3(16, HH, BB), 128, 0, stream>>>(Qb, Kb, Vt, M, AO);
  k_oproj <<<dim3(32, 256),     32, 0, stream>>>(AO, ow, ob, out);
}
